// MultiHeadAttention_55362128445825
// MI455X (gfx1250) — matrix operand order repaired and checked by reading, not compiled, not run on hardware
//
#include <hip/hip_runtime.h>
#include <hip/hip_bf16.h>

// MI455X / gfx1250: wave32, WMMA 16x16x32 bf16 -> f32 accumulate.

typedef __attribute__((ext_vector_type(16))) __bf16 v16bf;
typedef __attribute__((ext_vector_type(8)))  __bf16 v8bf;
typedef __attribute__((ext_vector_type(8)))  float  v8f;

#define WMMA_BF16(a, b, c) \
    __builtin_amdgcn_wmma_f32_16x16x32_bf16(false, (a), false, (b), (short)0, (c), false, false)

// Wave-local LDS producer->consumer fence: all DS ops from this wave complete.
#define WAVE_LDS_FENCE() asm volatile("s_wait_dscnt 0x0" ::: "memory")

static __device__ inline v16bf join8(v8bf lo, v8bf hi) {
    v16bf r;
#pragma unroll
    for (int i = 0; i < 8; ++i) { r[i] = lo[i]; r[8 + i] = hi[i]; }
    return r;
}

// ---------------------------------------------------------------------------
// GEMM: Y[m,n] = sum_k A[m,k] * W[n,k] + bias[n]   (torch Linear: x @ W.T + b)
// A: fp32 [4096 x 1024] row-major, W: fp32 [1024 x 1024] row-major.
// MODE 0: store bf16 head-major [B=2][H=16][S=2048][dk=64], scaled by `scale`.
// MODE 1: store fp32 row-major [4096 x 1024] (final projection).
// Block: 128 threads (4 waves). Tile: 128(M) x 64(N), K step 32.
// Wave w owns rows m0 + 32w .. +31 (two 16-row A-frags), all four N subtiles:
// 8 WMMAs per K-step per wave, amortizing the 4 LDS B-fragment loads.
// ---------------------------------------------------------------------------
template <int MODE>
__global__ __launch_bounds__(128) void gemm_bias_kernel(
    const float* __restrict__ A, const float* __restrict__ W,
    const float* __restrict__ bias, __bf16* __restrict__ out_bf16_hm,
    float* __restrict__ out_f32, float scale)
{
    __shared__ __align__(128) __bf16 Wt[32][64];  // W tile, transposed: [k][n], bf16

    const int t    = threadIdx.x;
    const int lane = t & 31;
    const int wave = t >> 5;
    const int m0 = blockIdx.x * 128;
    const int n0 = blockIdx.y * 64;

    v8f acc[2][4];
#pragma unroll
    for (int rg = 0; rg < 2; ++rg)
#pragma unroll
        for (int i = 0; i < 4; ++i) acc[rg][i] = v8f{};

    const int rA   = m0 + wave * 32 + (lane & 15);     // A-frag row (group 0)
    const int koff = (lane >> 4) * 8;                  // A-frag K sub-offset

    for (int kb = 0; kb < 1024; kb += 32) {
        // ---- stage W tile (64 n x 32 k) into LDS transposed, fp32 -> bf16 ----
        {
            const int nl = t & 63;
            const int ks = (t >> 6) * 16;
            const float* wp = W + (size_t)(n0 + nl) * 1024 + kb + ks;
            float wv[16];
            *(float4*)&wv[0]  = *(const float4*)(wp + 0);
            *(float4*)&wv[4]  = *(const float4*)(wp + 4);
            *(float4*)&wv[8]  = *(const float4*)(wp + 8);
            *(float4*)&wv[12] = *(const float4*)(wp + 12);
#pragma unroll
            for (int j = 0; j < 16; ++j) Wt[ks + j][nl] = (__bf16)wv[j];
        }
        __syncthreads();

        // ---- A fragments (2 x 16x32), fp32 global -> bf16 regs ----
        v16bf afr[2];
#pragma unroll
        for (int rg = 0; rg < 2; ++rg) {
            const float* ap = A + (size_t)(rA + rg * 16) * 1024 + kb + koff;
            float av[16];
            *(float4*)&av[0]  = *(const float4*)(ap + 0);
            *(float4*)&av[4]  = *(const float4*)(ap + 4);
            *(float4*)&av[8]  = *(const float4*)(ap + 16);
            *(float4*)&av[12] = *(const float4*)(ap + 20);
#pragma unroll
            for (int j = 0; j < 16; ++j) afr[rg][j] = (__bf16)av[j];
        }

#pragma unroll
        for (int ns = 0; ns < 4; ++ns) {
            v16bf bfr;   // B col N=lane&15, 16 K in table order
#pragma unroll
            for (int j = 0; j < 16; ++j) bfr[j] = Wt[koff + (j & 7) + ((j >> 3) << 4)][ns * 16 + (lane & 15)];
            acc[0][ns] = WMMA_BF16(afr[0], bfr, acc[0][ns]);
            acc[1][ns] = WMMA_BF16(afr[1], bfr, acc[1][ns]);
        }
        __syncthreads();
    }

    // ---- epilogue: +bias, *scale, store (branch-free, templated) ----
#pragma unroll
    for (int rg = 0; rg < 2; ++rg) {
#pragma unroll
        for (int ns = 0; ns < 4; ++ns) {
            const int n  = n0 + ns * 16 + (lane & 15);
            const float bv = bias[n];
#pragma unroll
            for (int g = 0; g < 8; ++g) {
                const int m = m0 + wave * 32 + rg * 16 + g + 8 * (lane >> 4);
                const float v = (acc[rg][ns][g] + bv) * scale;
                if (MODE == 1) {
                    out_f32[(size_t)m * 1024 + n] = v;
                } else {
                    const int b = m >> 11, s = m & 2047;     // S = 2048
                    const int h = n >> 6,  d = n & 63;       // dk = 64
                    out_bf16_hm[(((size_t)(b * 16 + h) * 2048) + s) * 64 + d] = (__bf16)v;
                }
            }
        }
    }
}

// ---------------------------------------------------------------------------
// Flash-style attention, transposed-score formulation: S^T = K @ Q^T.
// qh/kh/vh: bf16 [BH=32][S=2048][dk=64]; qh pre-scaled by 1/sqrt(dk).
// Output ao: fp32 [B*S=4096][1024] (heads re-interleaved).
// Grid: (BH=32, S/64=32), 128 threads (4 waves). Each wave owns a 16-row Q
// tile. Q^T is staged ONCE into a wave-private LDS patch and held resident as
// two B-fragments; K A-fragments stream straight from global (row-major, no
// staging). All LDS regions are wave-private -> NO workgroup barriers; only
// wave-local s_wait_dscnt fences. Softmax stats are one scalar (m,l) per
// thread (per q column), combined across the two lane halves via shfl_xor(16).
// ---------------------------------------------------------------------------
__global__ __launch_bounds__(128) void attn_kernel(
    const __bf16* __restrict__ qh, const __bf16* __restrict__ kh,
    const __bf16* __restrict__ vh, float* __restrict__ ao)
{
    __shared__ __align__(128) __bf16 QT[4][64][16];   // per-wave Q^T: [dk][q], 8 KB
    __shared__ __align__(128) __bf16 Pst[4][16][32];  // per-wave P tile: [q][key], 4 KB

    const int t    = threadIdx.x;
    const int lane = t & 31;
    const int wave = t >> 5;
    const int bh = blockIdx.x;            // 0..31
    const int b  = bh >> 4, h = bh & 15;
    const int q0 = blockIdx.y * 64 + wave * 16;

    const __bf16* Q = qh + (size_t)bh * 2048 * 64;
    const __bf16* K = kh + (size_t)bh * 2048 * 64;
    const __bf16* V = vh + (size_t)bh * 2048 * 64;

    const int r    = lane & 15;           // A-frag row / P row
    const int koff = (lane >> 4) * 8;     // A-frag K sub-offset

    // ---- stage Q^T once (wave-local): lane covers q = lane/2, half of dk ----
    {
        const int qrow = lane >> 1;
        const int dseg = (lane & 1) * 32;
        const __bf16* qp = Q + (size_t)(q0 + qrow) * 64 + dseg;
        v8bf a = *(const v8bf*)(qp + 0);
        v8bf c = *(const v8bf*)(qp + 8);
        v8bf e = *(const v8bf*)(qp + 16);
        v8bf f = *(const v8bf*)(qp + 24);
#pragma unroll
        for (int j = 0; j < 8; ++j) {
            QT[wave][dseg + j][qrow]      = a[j];
            QT[wave][dseg + 8 + j][qrow]  = c[j];
            QT[wave][dseg + 16 + j][qrow] = e[j];
            QT[wave][dseg + 24 + j][qrow] = f[j];
        }
    }
    WAVE_LDS_FENCE();

    // Resident B-fragments of Q^T: lane&15 = q column, elements = 16 dk rows (table order).
    v16bf bq[2];
#pragma unroll
    for (int ks = 0; ks < 2; ++ks)
#pragma unroll
        for (int j = 0; j < 16; ++j)
            bq[ks][j] = QT[wave][ks * 32 + koff + (j & 7) + ((j >> 3) << 4)][r];

    v8f o[4];
#pragma unroll
    for (int i = 0; i < 4; ++i) o[i] = v8f{};
    float m = -__builtin_inff();
    float l = 0.0f;

    for (int kb = 0; kb < 64; ++kb) {     // key blocks of 32
        const int kbase = kb * 32;

        // Prefetch next K/V block into cache hierarchy (global_prefetch_b8).
        if (kb + 1 < 64) {
            __builtin_prefetch(K + (size_t)(kbase + 32 + lane) * 64, 0, 3);
            __builtin_prefetch(V + (size_t)(kbase + 32 + lane) * 64, 0, 3);
        }

        // ---- scores^T: two 16-key tiles; A = K rows (global), B = Q^T (regs)
        v8f c[2];
#pragma unroll
        for (int ct = 0; ct < 2; ++ct) {
            const __bf16* kp = K + (size_t)(kbase + ct * 16 + r) * 64 + koff;
            v16bf a0 = join8(*(const v8bf*)(kp + 0),  *(const v8bf*)(kp + 16));
            v16bf a1 = join8(*(const v8bf*)(kp + 32), *(const v8bf*)(kp + 48));
            c[ct] = v8f{};
            c[ct] = WMMA_BF16(a0, bq[0], c[ct]);
            c[ct] = WMMA_BF16(a1, bq[1], c[ct]);
        }

        // ---- online softmax over key dim (C rows). Thread owns q = lane&15;
        //      its 16 keys + partner half's 16 keys via shfl_xor(16).
        float tmax = c[0][0];
#pragma unroll
        for (int ct = 0; ct < 2; ++ct)
#pragma unroll
            for (int g = 0; g < 8; ++g) tmax = fmaxf(tmax, c[ct][g]);
        tmax = fmaxf(tmax, __shfl_xor(tmax, 16, 32));
        const float mnew  = fmaxf(m, tmax);
        const float alpha = __expf(m - mnew);

        float psum = 0.0f;
#pragma unroll
        for (int ct = 0; ct < 2; ++ct)
#pragma unroll
            for (int g = 0; g < 8; ++g) {
                const float p = __expf(c[ct][g] - mnew);
                psum += p;
                Pst[wave][lane & 15][ct * 16 + g + 8 * (lane >> 4)] = (__bf16)p;
            }
        psum += __shfl_xor(psum, 16, 32);
        l = l * alpha + psum;
        m = mnew;

        // Rescale O accumulators: row q' = g + 8*(lane>>4) needs alpha[q'].
#pragma unroll
        for (int g = 0; g < 8; ++g) {
            const float ag = __shfl(alpha, g + 8 * (lane >> 4), 32);
#pragma unroll
            for (int sub = 0; sub < 4; ++sub) o[sub][g] *= ag;
        }

        WAVE_LDS_FENCE();   // P stores visible wave-wide

        // ---- PV: O[16 q x 64 dk] += P[16 q x 32 key] @ V[32 key x 64 dk] ----
        const __bf16* pp = &Pst[wave][r][koff];
        v16bf apf = join8(*(const v8bf*)pp, *(const v8bf*)(pp + 16));
#pragma unroll
        for (int sub = 0; sub < 4; ++sub) {
            v16bf bv;
#pragma unroll
            for (int j = 0; j < 16; ++j)
                bv[j] = V[(size_t)(kbase + koff + (j & 7) + ((j >> 3) << 4)) * 64 + sub * 16 + r];
            o[sub] = WMMA_BF16(apf, bv, o[sub]);
        }
    }

    // ---- epilogue: normalize by row sum (broadcast via shfl), store heads ----
#pragma unroll
    for (int g = 0; g < 8; ++g) {
        const int qp   = g + 8 * (lane >> 4);
        const float lr = __shfl(l, qp, 32);
        const int srow = blockIdx.y * 64 + wave * 16 + qp;
#pragma unroll
        for (int sub = 0; sub < 4; ++sub) {
            const int col = h * 64 + sub * 16 + (lane & 15);
            ao[((size_t)b * 2048 + srow) * 1024 + col] = o[sub][g] / lr;
        }
    }
}

// ---------------------------------------------------------------------------
// Host launcher. Inputs: q,k,v,Wq,bq,Wk,bk,Wv,bv,Wo,bo (fp32).
// Workspace layout (40 MB): qh(8MB bf16) | kh(8MB) | vh(8MB) | ao(16MB fp32).
// ---------------------------------------------------------------------------
extern "C" void kernel_launch(void* const* d_in, const int* in_sizes, int n_in,
                              void* d_out, int out_size, void* d_ws, size_t ws_size,
                              hipStream_t stream) {
    const float* q  = (const float*)d_in[0];
    const float* k  = (const float*)d_in[1];
    const float* v  = (const float*)d_in[2];
    const float* Wq = (const float*)d_in[3];
    const float* bq = (const float*)d_in[4];
    const float* Wk = (const float*)d_in[5];
    const float* bk = (const float*)d_in[6];
    const float* Wv = (const float*)d_in[7];
    const float* bv = (const float*)d_in[8];
    const float* Wo = (const float*)d_in[9];
    const float* bo = (const float*)d_in[10];

    char* ws = (char*)d_ws;
    __bf16* qh = (__bf16*)(ws);
    __bf16* kh = (__bf16*)(ws + (size_t)8  * 1024 * 1024);
    __bf16* vh = (__bf16*)(ws + (size_t)16 * 1024 * 1024);
    float*  ao = (float*) (ws + (size_t)24 * 1024 * 1024);

    const dim3 ggemm(32, 16);           // M/128 x N/64
    const float inv_sqrt_dk = 0.125f;   // 1/sqrt(64), folded into Q projection

    gemm_bias_kernel<0><<<ggemm, 128, 0, stream>>>(q, Wq, bq, qh, nullptr, inv_sqrt_dk);
    gemm_bias_kernel<0><<<ggemm, 128, 0, stream>>>(k, Wk, bk, kh, nullptr, 1.0f);
    gemm_bias_kernel<0><<<ggemm, 128, 0, stream>>>(v, Wv, bv, vh, nullptr, 1.0f);

    attn_kernel<<<dim3(32, 32), 128, 0, stream>>>(qh, kh, vh, ao);

    gemm_bias_kernel<1><<<ggemm, 128, 0, stream>>>(ao, Wo, bo, nullptr, (float*)d_out, 1.0f);
}
